// se_BiLSTM_6193342841427
// MI455X (gfx1250) — matrix operand order repaired and checked by reading, not compiled, not run on hardware
//
#include <hip/hip_runtime.h>
#include <math.h>

typedef __bf16 bf16_t;
typedef __attribute__((ext_vector_type(16))) __bf16 v16bf;
typedef __attribute__((ext_vector_type(8)))  __bf16 v8bf;
typedef __attribute__((ext_vector_type(8)))  float  v8f;

#define B_   64
#define T_   1024
#define D_   512
#define U_   256
#define G4_  1024            /* 4 * UNITS */
#define M_   (B_ * T_)       /* 65536 GEMM rows, m = t*64 + b */

__device__ __forceinline__ v16bf pack16(v8bf lo, v8bf hi) {
  v16bf r;
#pragma unroll
  for (int i = 0; i < 8; ++i) { r[i] = lo[i]; r[i + 8] = hi[i]; }
  return r;
}

// A fragment (16x32 bf16). Per ISA: lanes 0-15 row M=lane, K runs {k0..k0+7, k0+16..k0+23};
// lanes 16-31 same rows, K runs shifted by 8.
__device__ __forceinline__ v16bf load_a_frag(const bf16_t* rowPtr, int k0, int half) {
  v8bf lo = *(const v8bf*)(rowPtr + k0 + 8 * half);
  v8bf hi = *(const v8bf*)(rowPtr + k0 + 16 + 8 * half);
  return pack16(lo, hi);
}

// B fragment (32x16 bf16) from column-major source (Ut[n][k]): lane col N=lane%16,
// K runs {k0+8*(lane/16)..+7, k0+16+8*(lane/16)..+7} (same as A).
__device__ __forceinline__ v16bf load_b_frag(const bf16_t* colPtr, int k0, int half) {
  const bf16_t* p = colPtr + k0 + 8 * half;
  v8bf lo = *(const v8bf*)(p);
  v8bf hi = *(const v8bf*)(p + 16);
  return pack16(lo, hi);
}

__device__ __forceinline__ v8f wmma_bf16(v16bf a, v16bf b, v8f c) {
  return __builtin_amdgcn_wmma_f32_16x16x32_bf16(
      /*neg_a=*/false, a, /*neg_b=*/false, b,
      /*c_mod=*/(short)0, c, /*reuse_a=*/false, /*reuse_b=*/false);
}

// xz is stored in C/D fragment layout: [t][col16][mt][lane][r] bf16, where
// col16 = (gate column)/16 in 0..63, mt = batch 16-row subtile, lane 0..31,
// r = accumulator VGPR index 0..7. One lane's 8 elements are contiguous (16B).
__device__ __forceinline__ size_t xz_off(int t, int col16, int mt, int lane) {
  return ((((size_t)t * 64 + col16) * 4 + mt) * 32 + lane) * 8;
}

// ---------------------------------------------------------------------------
// Kernel 1: precision prep. x -> bf16 (layout kept [B][T][D]);
// W (D x 4U) -> Wt bf16 [dir][n][d]; U (U x 4U) -> Ut bf16 [dir][n][k].
// ---------------------------------------------------------------------------
__global__ void bilstm_prep(const float* __restrict__ x,
                            const float* __restrict__ Wf, const float* __restrict__ Wb,
                            const float* __restrict__ Uf, const float* __restrict__ Ub,
                            bf16_t* __restrict__ xb, bf16_t* __restrict__ Wt,
                            bf16_t* __restrict__ Ut) {
  const size_t NX = (size_t)B_ * T_ * D_;        // 33554432
  const size_t NW = (size_t)G4_ * D_;            // 524288 per dir
  const size_t NU = (size_t)G4_ * U_;            // 262144 per dir
  const size_t TOTAL = NX + 2 * NW + 2 * NU;
  const size_t stride = (size_t)gridDim.x * blockDim.x;
  for (size_t idx = (size_t)blockIdx.x * blockDim.x + threadIdx.x; idx < TOTAL; idx += stride) {
    if (idx < NX) {
      xb[idx] = (bf16_t)x[idx];
    } else if (idx < NX + 2 * NW) {
      size_t j = idx - NX;
      int dir = (int)(j / NW);
      size_t r = j - (size_t)dir * NW;
      size_t n = r / D_, d = r % D_;
      const float* W = dir ? Wb : Wf;
      Wt[j] = (bf16_t)W[d * G4_ + n];
    } else {
      size_t j = idx - NX - 2 * NW;
      int dir = (int)(j / NU);
      size_t r = j - (size_t)dir * NU;
      size_t n = r / U_, k = r % U_;
      const float* U = dir ? Ub : Uf;
      Ut[j] = (bf16_t)U[k * G4_ + n];
    }
  }
}

// ---------------------------------------------------------------------------
// Kernel 2: xz = x @ W + b (bf16, fragment layout). m = t*64 + b, so a 64-row
// m-tile is exactly one timestep t = mT. Each wave: 64x64 tile, K=512.
// ---------------------------------------------------------------------------
__global__ void __launch_bounds__(256) bilstm_xproj(
    const bf16_t* __restrict__ xb,    // [64][1024][512]
    const bf16_t* __restrict__ Wt,    // [2][1024][512]
    const float* __restrict__ bf_, const float* __restrict__ bb_,
    bf16_t* __restrict__ xz)          // [2][T][64col16][4mt][32lane][8r]
{
  const int dir = blockIdx.z;
  const bf16_t* Wd   = Wt + (size_t)dir * G4_ * D_;
  const float*  bias = dir ? bb_ : bf_;
  bf16_t*       xzd  = xz + (size_t)dir * M_ * G4_;

  const int lane = threadIdx.x & 31;
  const int wave = threadIdx.x >> 5;
  const int tile = blockIdx.x * 8 + wave;   // 16384 tiles of 64x64
  const int nT = tile & 15;                 // 16 n-tiles of 64
  const int mT = tile >> 4;                 // 1024 m-tiles of 64 == timestep t
  const int half = lane >> 4;
  const int l16  = lane & 15;

  const bf16_t* arow[4];
#pragma unroll
  for (int mt = 0; mt < 4; ++mt) {
    int m = mT * 64 + mt * 16 + l16;        // m = t*64 + b
    int b = m & 63, t = m >> 6;
    arow[mt] = xb + ((size_t)b * T_ + t) * D_;
  }
  const bf16_t* bcol[4];
#pragma unroll
  for (int nt = 0; nt < 4; ++nt) {
    int n = nT * 64 + nt * 16 + l16;
    bcol[nt] = Wd + (size_t)n * D_;
  }

  v8f acc[4][4];
#pragma unroll
  for (int mt = 0; mt < 4; ++mt)
#pragma unroll
    for (int nt = 0; nt < 4; ++nt)
#pragma unroll
      for (int r = 0; r < 8; ++r) acc[mt][nt][r] = 0.0f;

  for (int ks = 0; ks < 16; ++ks) {
    const int k0 = ks * 32;
    v16bf af[4];
#pragma unroll
    for (int mt = 0; mt < 4; ++mt) af[mt] = load_a_frag(arow[mt], k0, half);
#pragma unroll
    for (int nt = 0; nt < 4; ++nt) {
      v16bf bfr = load_b_frag(bcol[nt], k0, half);
#pragma unroll
      for (int mt = 0; mt < 4; ++mt)
        acc[mt][nt] = wmma_bf16(af[mt], bfr, acc[mt][nt]);
    }
  }

  // epilogue: add bias, convert to bf16, one 16B non-temporal store per tile
#pragma unroll
  for (int nt = 0; nt < 4; ++nt) {
    const int col16 = nT * 4 + nt;
    const float bv = bias[col16 * 16 + l16];
#pragma unroll
    for (int mt = 0; mt < 4; ++mt) {
      v8bf s;
#pragma unroll
      for (int r = 0; r < 8; ++r) s[r] = (bf16_t)(acc[mt][nt][r] + bv);
      __builtin_nontemporal_store(s, (v8bf*)(xzd + xz_off(mT, col16, mt, lane)));
    }
  }
}

// ---------------------------------------------------------------------------
// Kernel 3: persistent recurrence, one block per direction, 16 waves.
// Wave w owns units [16w,16w+16): gate columns g*256+16w (col16 = g*16+w),
// so i/f/g/o accumulator tiles are element-aligned for the cell update.
// z accumulators are seeded from xz_t fragments via the WMMA C operand.
// ---------------------------------------------------------------------------
__global__ void __launch_bounds__(512) bilstm_recur(
    const bf16_t* __restrict__ Ut,    // [2][1024][256] (transposed, bf16)
    const bf16_t* __restrict__ xz,    // [2] fragment layout
    float* __restrict__ hfin)         // [2][64][256]
{
  const int dir = blockIdx.x;
  const bf16_t* Ud  = Ut + (size_t)dir * G4_ * U_;
  const bf16_t* xzd = xz + (size_t)dir * M_ * G4_;
  float* hout = hfin + (size_t)dir * B_ * U_;

  __shared__ __align__(16) bf16_t hlds[B_ * U_];   // 32 KB: h state, bf16

  const int tid  = threadIdx.x;
  const int lane = tid & 31, wave = tid >> 5;
  const int half = lane >> 4, l16 = lane & 15;
  const int ubase = wave * 16;

  for (int i = tid; i < B_ * U_; i += 512) hlds[i] = (bf16_t)0.0f;
  __syncthreads();

  float creg[4][8];
#pragma unroll
  for (int mt = 0; mt < 4; ++mt)
#pragma unroll
    for (int r = 0; r < 8; ++r) creg[mt][r] = 0.0f;

  const bf16_t* bcol[4];
#pragma unroll
  for (int g = 0; g < 4; ++g)
    bcol[g] = Ud + (size_t)(g * 256 + ubase + l16) * U_;

  // Warm WGP$/L2 with this wave's U columns (resident across all 1024 steps).
#pragma unroll
  for (int g = 0; g < 4; ++g)
#pragma unroll
    for (int off = 0; off < U_; off += 64)
      __builtin_prefetch((const void*)(bcol[g] + off), 0, 3);

  for (int step = 0; step < T_; ++step) {
    const int t = dir ? (T_ - 1 - step) : step;

    // Prefetch next timestep's xz fragments: independent of the h-chain, so
    // the HBM fetch overlaps this step's WMMAs + gate math.
    if (step + 1 < T_) {
      const int tn = dir ? (T_ - 2 - step) : (step + 1);
#pragma unroll
      for (int g = 0; g < 4; ++g) {
        const int col16 = g * 16 + wave;
#pragma unroll
        for (int mt = 0; mt < 4; ++mt)
          __builtin_prefetch((const void*)(xzd + xz_off(tn, col16, mt, lane)), 0, 1);
      }
    }

    // seed accumulators with xz_t fragments: one 16B NT load per tile
    v8f acc[4][4];  // [mt][gate]
#pragma unroll
    for (int g = 0; g < 4; ++g) {
      const int col16 = g * 16 + wave;
#pragma unroll
      for (int mt = 0; mt < 4; ++mt) {
        v8bf s = __builtin_nontemporal_load(
            (const v8bf*)(xzd + xz_off(t, col16, mt, lane)));
#pragma unroll
        for (int r = 0; r < 8; ++r) acc[mt][g][r] = (float)s[r];
      }
    }

    // z += h @ U   (M=64 batch, N=64 owned cols, K=256)
#pragma unroll
    for (int ks = 0; ks < 8; ++ks) {
      const int k0 = ks * 32;
      v16bf af[4];
#pragma unroll
      for (int mt = 0; mt < 4; ++mt)
        af[mt] = load_a_frag(hlds + (size_t)(mt * 16 + l16) * U_, k0, half);
#pragma unroll
      for (int g = 0; g < 4; ++g) {
        v16bf bfr = load_b_frag(bcol[g], k0, half);
#pragma unroll
        for (int mt = 0; mt < 4; ++mt)
          acc[mt][g] = wmma_bf16(af[mt], bfr, acc[mt][g]);
      }
    }
    __syncthreads();  // all waves done reading hlds

    // gate math + cell update; write next h to LDS
    const int u = ubase + l16;
#pragma unroll
    for (int mt = 0; mt < 4; ++mt) {
#pragma unroll
      for (int r = 0; r < 8; ++r) {
        const int b = mt * 16 + 8 * half + r;
        float gi = tanhf(acc[mt][0][r]);   // recurrent_activation = tanh
        float gf = tanhf(acc[mt][1][r]);
        float gg = tanhf(acc[mt][2][r]);
        float go = tanhf(acc[mt][3][r]);
        float c  = gf * creg[mt][r] + gi * gg;
        creg[mt][r] = c;
        float h = go * tanhf(c);
        hlds[b * U_ + u] = (bf16_t)h;
        if (step == T_ - 1) hout[b * U_ + u] = h;
      }
    }
    __syncthreads();  // h visible before next step's A-frag loads
  }
}

// ---------------------------------------------------------------------------
// Kernel 4: out = h_fwd + h_bwd
// ---------------------------------------------------------------------------
__global__ void bilstm_combine(const float* __restrict__ hfin, float* __restrict__ out) {
  int i = blockIdx.x * blockDim.x + threadIdx.x;
  if (i < B_ * U_) out[i] = hfin[i] + hfin[B_ * U_ + i];
}

// ---------------------------------------------------------------------------
extern "C" void kernel_launch(void* const* d_in, const int* in_sizes, int n_in,
                              void* d_out, int out_size, void* d_ws, size_t ws_size,
                              hipStream_t stream) {
  const float* x  = (const float*)d_in[0];
  const float* Wf = (const float*)d_in[1];
  const float* Uf = (const float*)d_in[2];
  const float* bf = (const float*)d_in[3];
  const float* Wb = (const float*)d_in[4];
  const float* Ub = (const float*)d_in[5];
  const float* bb = (const float*)d_in[6];
  float* out = (float*)d_out;

  // workspace carve (all offsets 256B-aligned)
  char* ws = (char*)d_ws;
  size_t off = 0;
  bf16_t* xb = (bf16_t*)(ws + off); off += (size_t)B_ * T_ * D_ * 2;        // 67,108,864
  bf16_t* Wt = (bf16_t*)(ws + off); off += 2ull * G4_ * D_ * 2;             //  2,097,152
  bf16_t* Ut = (bf16_t*)(ws + off); off += 2ull * G4_ * U_ * 2;             //  1,048,576
  bf16_t* xz = (bf16_t*)(ws + off); off += 2ull * M_ * G4_ * 2;             // 268,435,456
  float* hfin = (float*)(ws + off); off += 2ull * B_ * U_ * 4;              //    131,072
  (void)ws_size; (void)in_sizes; (void)n_in; (void)out_size;

  // 1) fp32 -> bf16 + W/U transposition
  bilstm_prep<<<4096, 256, 0, stream>>>(x, Wf, Wb, Uf, Ub, xb, Wt, Ut);

  // 2) big input-projection GEMM, both directions (z dim), 8 wave-tiles/block
  bilstm_xproj<<<dim3(2048, 1, 2), 256, 0, stream>>>(xb, Wt, bf, bb, xz);

  // 3) persistent recurrence: block 0 = forward, block 1 = backward
  bilstm_recur<<<2, 512, 0, stream>>>(Ut, xz, hfin);

  // 4) h_f + h_b -> out [64,256]
  bilstm_combine<<<64, 256, 0, stream>>>(hfin, out);
}
